// MultiHeadNeuroAttention_68856915689657
// MI455X (gfx1250) — matrix operand order repaired and checked by reading, not compiled, not run on hardware
//
#include <hip/hip_runtime.h>

#define L_SEQ  4096
#define DMODEL 512
#define HEADS  8
#define DH     64

typedef unsigned short ushort_t;
typedef __attribute__((ext_vector_type(16))) __bf16 v16bf;
typedef __attribute__((ext_vector_type(8)))  float  v8f;

union FragU { v16bf v; uint4 q[2]; };

__device__ __forceinline__ ushort_t f32_to_bf16(float f) {
  unsigned int u = __float_as_uint(f);
  u += 0x7fffu + ((u >> 16) & 1u);   // round to nearest even
  return (ushort_t)(u >> 16);
}

__device__ __forceinline__ v8f v8f_zero() {
  v8f z;
#pragma unroll
  for (int i = 0; i < 8; ++i) z[i] = 0.0f;
  return z;
}

// A fragment (16x32, MxK) from row-major bf16 LDS tile.
// lane<16:  m=lane,    elems0-7 K=kslice+0..7,  elems8-15 K=kslice+16..23
// lane>=16: m=lane-16, elems0-7 K=kslice+8..15, elems8-15 K=kslice+24..31
__device__ __forceinline__ v16bf load_frag_a(const ushort_t* base, int row0,
                                             int stride, int kslice, int lane) {
  const int m  = lane & 15;
  const int hi = (lane >> 4) & 1;
  const ushort_t* p = base + (row0 + m) * stride + kslice + hi * 8;
  FragU f;
  f.q[0] = *(const uint4*)(p);
  f.q[1] = *(const uint4*)(p + 16);
  return f.v;
}

// B fragment (32x16, KxN) from K-contiguous transposed LDS tile [n][k].
// lane holds column n=lane&15; elems0-7 K = kslice + (lane>=16?8:0) + e, elems8-15 K = kslice + 16 + (lane>=16?8:0) + (e-8)
__device__ __forceinline__ v16bf load_frag_b(const ushort_t* baseT, int col0,
                                             int stride, int kslice, int lane) {
  const int n  = lane & 15;
  const int kb = ((lane >> 4) & 1) * 8;
  const ushort_t* p = baseT + (col0 + n) * stride + kslice + kb;
  FragU f;
  f.q[0] = *(const uint4*)(p);
  f.q[1] = *(const uint4*)(p + 16);
  return f.v;
}

#define WMMA_BF16(a, b, c) \
  __builtin_amdgcn_wmma_f32_16x16x32_bf16(false, (a), false, (b), (short)0, (c), false, false)

// ---------------- elementwise f32 -> bf16 ----------------
__global__ void cvt_bf16_kernel(const float* __restrict__ in,
                                ushort_t* __restrict__ out, int n) {
  int i = blockIdx.x * blockDim.x + threadIdx.x;
  if (i < n) out[i] = f32_to_bf16(in[i]);
}

// ---------------- fuse two-stage projection weights ----------------
// Weff[i][j] = sum_k W1[i][k]*W2[k][j]  (512x512x512), output bf16
__global__ __launch_bounds__(256) void fuse_weight_kernel(
    const float* __restrict__ W1, const float* __restrict__ W2,
    ushort_t* __restrict__ Weff) {
  const int i = blockIdx.x;
  const int j = blockIdx.y * 256 + threadIdx.x;
  float s = 0.0f;
  for (int k = 0; k < 512; ++k) s = fmaf(W1[i * 512 + k], W2[k * 512 + j], s);
  Weff[i * 512 + j] = f32_to_bf16(s);
}

// beff[j] = sum_k b1[k]*W2[k][j] + b2[j]
__global__ __launch_bounds__(256) void fuse_bias_kernel(
    const float* __restrict__ b1, const float* __restrict__ W2,
    const float* __restrict__ b2, float* __restrict__ beff) {
  const int j = blockIdx.x * 256 + threadIdx.x;
  float s = b2[j];
  for (int k = 0; k < 512; ++k) s = fmaf(b1[k], W2[k * 512 + j], s);
  beff[j] = s;
}

// ---------------- bf16 WMMA GEMM: C = A(bf16)@B(bf16) + bias ----------------
// 128x128 block tile, BK=32, 8 waves each computing a 64x32 macro-tile.
template <bool OUT_F32>
__global__ __launch_bounds__(256) void gemm_bf16_kernel(
    const ushort_t* __restrict__ A, const ushort_t* __restrict__ B,
    const float* __restrict__ bias, void* __restrict__ Cout,
    int M, int N, int K) {
  __shared__ ushort_t As[128 * 32];   // [row][k]
  __shared__ ushort_t Bt[128 * 32];   // [n][k] (transposed)
  const int tid  = threadIdx.x;
  const int lane = tid & 31;
  const int wv   = tid >> 5;
  const int bm = blockIdx.y * 128;
  const int bn = blockIdx.x * 128;
  const int wr = (wv & 1) * 64;
  const int wc = (wv >> 1) * 32;

  v8f acc[4][2];
#pragma unroll
  for (int i = 0; i < 4; ++i)
#pragma unroll
    for (int j = 0; j < 2; ++j) acc[i][j] = v8f_zero();

  for (int k0 = 0; k0 < K; k0 += 32) {
    {  // A tile 128x32, 16 bf16 per thread
      const int r = tid >> 1, cg = (tid & 1) * 16;
      const uint4* g = (const uint4*)(A + (size_t)(bm + r) * K + k0 + cg);
      *(uint4*)&As[r * 32 + cg]     = g[0];
      *(uint4*)&As[r * 32 + cg + 8] = g[1];
    }
    {  // B tile 32x128 loaded coalesced, stored transposed
      const int kk = tid >> 3, ng = (tid & 7) * 16;
      const ushort_t* g = B + (size_t)(k0 + kk) * N + bn + ng;
      __align__(16) ushort_t tmp[16];
      *(uint4*)&tmp[0] = *(const uint4*)(g);
      *(uint4*)&tmp[8] = *(const uint4*)(g + 8);
#pragma unroll
      for (int e = 0; e < 16; ++e) Bt[(ng + e) * 32 + kk] = tmp[e];
    }
    __syncthreads();
    v16bf afr[4], bfr[2];
#pragma unroll
    for (int i = 0; i < 4; ++i) afr[i] = load_frag_a(As, wr + i * 16, 32, 0, lane);
#pragma unroll
    for (int j = 0; j < 2; ++j) bfr[j] = load_frag_b(Bt, wc + j * 16, 32, 0, lane);
#pragma unroll
    for (int i = 0; i < 4; ++i)
#pragma unroll
      for (int j = 0; j < 2; ++j)
        acc[i][j] = WMMA_BF16(afr[i], bfr[j], acc[i][j]);
    __syncthreads();
  }

  const int hi = lane >> 4;
  const int n  = lane & 15;
#pragma unroll
  for (int i = 0; i < 4; ++i) {
#pragma unroll
    for (int j = 0; j < 2; ++j) {
      const int col = bn + wc + j * 16 + n;
      const float bv = bias[col];
#pragma unroll
      for (int v = 0; v < 8; ++v) {
        const int row   = bm + wr + i * 16 + v + hi * 8;
        const float val = acc[i][j][v] + bv;
        if (OUT_F32) ((float*)Cout)[(size_t)row * N + col] = val;
        else         ((ushort_t*)Cout)[(size_t)row * N + col] = f32_to_bf16(val);
      }
    }
  }
}

// ------------- materialize SP = 0.5*softmax(shortest_path, axis=-1) ---------
// Head-invariant: computed once, reused by all 8 heads (64MB, L2-resident).
__global__ __launch_bounds__(256) void sp_softmax_kernel(
    const float* __restrict__ sp, float* __restrict__ spn) {
  const int row = blockIdx.x;
  const float* p = sp + (size_t)row * L_SEQ;
  float* o = spn + (size_t)row * L_SEQ;
  __shared__ float red[256];
  float loc[16];
  float m = -1e30f;
#pragma unroll
  for (int i = 0; i < 16; ++i) {
    loc[i] = p[threadIdx.x + 256 * i];
    m = fmaxf(m, loc[i]);
  }
  red[threadIdx.x] = m;
  __syncthreads();
  for (int s = 128; s > 0; s >>= 1) {
    if (threadIdx.x < s) red[threadIdx.x] = fmaxf(red[threadIdx.x], red[threadIdx.x + s]);
    __syncthreads();
  }
  m = red[0];
  __syncthreads();
  float sum = 0.0f;
#pragma unroll
  for (int i = 0; i < 16; ++i) {
    loc[i] = __expf(loc[i] - m);
    sum += loc[i];
  }
  red[threadIdx.x] = sum;
  __syncthreads();
  for (int s = 128; s > 0; s >>= 1) {
    if (threadIdx.x < s) red[threadIdx.x] += red[threadIdx.x + s];
    __syncthreads();
  }
  const float inv = 0.5f / red[0];
#pragma unroll
  for (int i = 0; i < 16; ++i) o[threadIdx.x + 256 * i] = loc[i] * inv;
}

// ---------------- flash attention with precomputed SP bias -------------------
// Grid (head, qblock); head fastest-varying so consecutive blocks share SP rows
// in L2. Block = 8 waves; each wave owns 16 query rows; 64-key tiles.
__global__ __launch_bounds__(256) void flash_attn_kernel(
    const ushort_t* __restrict__ Q, const ushort_t* __restrict__ Kg,
    const ushort_t* __restrict__ Vg, const float* __restrict__ SP,
    ushort_t* __restrict__ O) {
  const int h   = blockIdx.x;
  const int q0  = blockIdx.y * 128;
  const int tid = threadIdx.x, lane = tid & 31, wv = tid >> 5;
  const int hi  = lane >> 4, nn = lane & 15;

  __shared__ ushort_t Qs[128 * 64];     // [qrow][f]
  __shared__ ushort_t Ks[64 * 64];      // [key][f]   (B^T layout for QK^T)
  __shared__ ushort_t Vt[64 * 64];      // [dv][key]  (B^T layout for PV)
  __shared__ ushort_t Ps[8][16 * 64];   // per-wave P staging (C->A relayout)

  {  // Q tile 128x64: 32 bf16 per thread
    const int r = tid >> 1, cg = (tid & 1) * 32;
    const uint4* g = (const uint4*)(Q + (size_t)(q0 + r) * DMODEL + h * DH + cg);
    uint4* s = (uint4*)&Qs[r * 64 + cg];
    s[0] = g[0]; s[1] = g[1]; s[2] = g[2]; s[3] = g[3];
  }

  v8f oc[4];
#pragma unroll
  for (int c = 0; c < 4; ++c) oc[c] = v8f_zero();
  float mrow[8], lrow[8];
  const float* spr[8];
#pragma unroll
  for (int v = 0; v < 8; ++v) {
    const int row = q0 + wv * 16 + v + hi * 8;
    mrow[v] = -1e30f;
    lrow[v] = 0.0f;
    spr[v]  = SP + (size_t)row * L_SEQ;
  }
  const float scale = 0.125f;  // 1/sqrt(64)

  for (int k0 = 0; k0 < L_SEQ; k0 += 64) {
    __syncthreads();  // previous iter's K/V reads done before overwrite
    {  // K tile 64x64, coalesced
      const int key = tid >> 2, fg = (tid & 3) * 16;
      const uint4* g = (const uint4*)(Kg + (size_t)(k0 + key) * DMODEL + h * DH + fg);
      uint4* s = (uint4*)&Ks[key * 64 + fg];
      s[0] = g[0]; s[1] = g[1];
    }
    {  // V tile 64x64 transposed; lane->key fastest keeps LDS stores ~2-way
      const int kk = tid & 63, ng = (tid >> 6) * 16;
      __align__(16) ushort_t tmp[16];
      const ushort_t* g = Vg + (size_t)(k0 + kk) * DMODEL + h * DH + ng;
      *(uint4*)&tmp[0] = *(const uint4*)(g);
      *(uint4*)&tmp[8] = *(const uint4*)(g + 8);
#pragma unroll
      for (int e = 0; e < 16; ++e) Vt[(ng + e) * 64 + kk] = tmp[e];
    }
    if (k0 + 64 < L_SEQ) {  // hide next tile's latency behind WMMA+softmax
      __builtin_prefetch(Kg + (size_t)(k0 + 64 + (tid >> 2)) * DMODEL + h * DH + (tid & 3) * 16, 0, 1);
      __builtin_prefetch(Vg + (size_t)(k0 + 64 + (tid & 63)) * DMODEL + h * DH + (tid >> 6) * 16, 0, 1);
#pragma unroll
      for (int v = 0; v < 8; ++v) __builtin_prefetch(spr[v] + k0 + 64 + nn, 0, 1);
    }
    __syncthreads();

    // S = q(16x64) @ k_tile^T(64x64): 4 col tiles x 2 chained K=32 WMMAs
    v16bf a0 = load_frag_a(Qs, wv * 16, 64, 0, lane);
    v16bf a1 = load_frag_a(Qs, wv * 16, 64, 32, lane);
    v8f s[4];
#pragma unroll
    for (int ct = 0; ct < 4; ++ct) {
      s[ct] = v8f_zero();
      s[ct] = WMMA_BF16(a0, load_frag_b(Ks, ct * 16, 64, 0, lane), s[ct]);
      s[ct] = WMMA_BF16(a1, load_frag_b(Ks, ct * 16, 64, 32, lane), s[ct]);
    }

    // scale + precomputed SP bias (single L2-hit load/element), row max
    float rm[8];
#pragma unroll
    for (int v = 0; v < 8; ++v) {
      const float b0 = spr[v][k0 + nn];
      const float b1 = spr[v][k0 + 16 + nn];
      const float b2 = spr[v][k0 + 32 + nn];
      const float b3 = spr[v][k0 + 48 + nn];
      s[0][v] = fmaf(s[0][v], scale, b0);
      s[1][v] = fmaf(s[1][v], scale, b1);
      s[2][v] = fmaf(s[2][v], scale, b2);
      s[3][v] = fmaf(s[3][v], scale, b3);
      rm[v] = fmaxf(fmaxf(s[0][v], s[1][v]), fmaxf(s[2][v], s[3][v]));
    }
#pragma unroll
    for (int off = 1; off < 16; off <<= 1)
#pragma unroll
      for (int v = 0; v < 8; ++v)
        rm[v] = fmaxf(rm[v], __shfl_xor(rm[v], off, 32));

    // online softmax update (one exp per score element)
    float psum[8];
#pragma unroll
    for (int v = 0; v < 8; ++v) {
      const float mn    = fmaxf(mrow[v], rm[v]);
      const float alpha = __expf(mrow[v] - mn);
      mrow[v] = mn;
      lrow[v] = lrow[v] * alpha;
#pragma unroll
      for (int c = 0; c < 4; ++c) oc[c][v] = oc[c][v] * alpha;
      s[0][v] = __expf(s[0][v] - mn);
      s[1][v] = __expf(s[1][v] - mn);
      s[2][v] = __expf(s[2][v] - mn);
      s[3][v] = __expf(s[3][v] - mn);
      psum[v] = (s[0][v] + s[1][v]) + (s[2][v] + s[3][v]);
    }
#pragma unroll
    for (int off = 1; off < 16; off <<= 1)
#pragma unroll
      for (int v = 0; v < 8; ++v)
        psum[v] += __shfl_xor(psum[v], off, 32);
#pragma unroll
    for (int v = 0; v < 8; ++v) lrow[v] += psum[v];

    // P: C-layout -> bf16 LDS -> A fragments (wave-private, DS in-order)
    ushort_t* pw = Ps[wv];
#pragma unroll
    for (int v = 0; v < 8; ++v) {
      const int rb = (v + hi * 8) * 64;
      pw[rb + nn]      = f32_to_bf16(s[0][v]);
      pw[rb + 16 + nn] = f32_to_bf16(s[1][v]);
      pw[rb + 32 + nn] = f32_to_bf16(s[2][v]);
      pw[rb + 48 + nn] = f32_to_bf16(s[3][v]);
    }
    v16bf pa0 = load_frag_a(pw, 0, 64, 0, lane);
    v16bf pa1 = load_frag_a(pw, 0, 64, 32, lane);
#pragma unroll
    for (int c = 0; c < 4; ++c) {
      oc[c] = WMMA_BF16(pa0, load_frag_b(Vt, c * 16, 64, 0, lane), oc[c]);
      oc[c] = WMMA_BF16(pa1, load_frag_b(Vt, c * 16, 64, 32, lane), oc[c]);
    }
  }

  // epilogue: normalize and emit bf16 attention output
#pragma unroll
  for (int v = 0; v < 8; ++v) {
    const int row   = q0 + wv * 16 + v + hi * 8;
    const float inv = 1.0f / lrow[v];
#pragma unroll
    for (int c = 0; c < 4; ++c)
      O[(size_t)row * DMODEL + h * DH + c * 16 + nn] = f32_to_bf16(oc[c][v] * inv);
  }
}

// ---------------------------------------------------------------------------
extern "C" void kernel_launch(void* const* d_in, const int* in_sizes, int n_in,
                              void* d_out, int out_size, void* d_ws, size_t ws_size,
                              hipStream_t stream) {
  (void)in_sizes; (void)n_in; (void)out_size; (void)ws_size;
  const float* x     = (const float*)d_in[0];
  const float* sp    = (const float*)d_in[1];
  const float* wq_w  = (const float*)d_in[2];
  const float* wq_b  = (const float*)d_in[3];
  const float* wk_w  = (const float*)d_in[4];
  const float* wk_b  = (const float*)d_in[5];
  const float* wv_w  = (const float*)d_in[6];
  const float* wv_b  = (const float*)d_in[7];
  const float* qp_w  = (const float*)d_in[8];
  const float* qp_b  = (const float*)d_in[9];
  const float* kp_w  = (const float*)d_in[10];
  const float* kp_b  = (const float*)d_in[11];
  const float* vp_w  = (const float*)d_in[12];
  const float* vp_b  = (const float*)d_in[13];
  const float* out_w = (const float*)d_in[14];
  const float* out_b = (const float*)d_in[15];

  char* ws = (char*)d_ws;
  size_t off = 0;
  auto alloc = [&](size_t bytes) -> char* {
    char* p = ws + off;
    off += (bytes + 255) & ~(size_t)255;
    return p;
  };
  ushort_t* xbf  = (ushort_t*)alloc((size_t)L_SEQ * DMODEL * 2);
  ushort_t* wqe  = (ushort_t*)alloc((size_t)512 * 512 * 2);
  ushort_t* wke  = (ushort_t*)alloc((size_t)512 * 512 * 2);
  ushort_t* wve  = (ushort_t*)alloc((size_t)512 * 512 * 2);
  ushort_t* owbf = (ushort_t*)alloc((size_t)512 * 512 * 2);
  float*    bqe  = (float*)alloc(512 * 4);
  float*    bke  = (float*)alloc(512 * 4);
  float*    bve  = (float*)alloc(512 * 4);
  ushort_t* Qbf  = (ushort_t*)alloc((size_t)L_SEQ * DMODEL * 2);
  ushort_t* Kbf  = (ushort_t*)alloc((size_t)L_SEQ * DMODEL * 2);
  ushort_t* Vbf  = (ushort_t*)alloc((size_t)L_SEQ * DMODEL * 2);
  ushort_t* Obf  = (ushort_t*)alloc((size_t)L_SEQ * DMODEL * 2);
  float*    SPn  = (float*)alloc((size_t)L_SEQ * L_SEQ * 4);  // 64MB, L2-resident

  const int nX = L_SEQ * DMODEL;
  cvt_bf16_kernel<<<(nX + 255) / 256, 256, 0, stream>>>(x, xbf, nX);
  cvt_bf16_kernel<<<(512 * 512 + 255) / 256, 256, 0, stream>>>(out_w, owbf, 512 * 512);

  fuse_weight_kernel<<<dim3(512, 2), 256, 0, stream>>>(wq_w, qp_w, wqe);
  fuse_weight_kernel<<<dim3(512, 2), 256, 0, stream>>>(wk_w, kp_w, wke);
  fuse_weight_kernel<<<dim3(512, 2), 256, 0, stream>>>(wv_w, vp_w, wve);
  fuse_bias_kernel<<<2, 256, 0, stream>>>(wq_b, qp_w, qp_b, bqe);
  fuse_bias_kernel<<<2, 256, 0, stream>>>(wk_b, kp_w, kp_b, bke);
  fuse_bias_kernel<<<2, 256, 0, stream>>>(wv_b, vp_w, vp_b, bve);

  // SP = 0.5*softmax(shortest_path) materialized once (head-invariant)
  sp_softmax_kernel<<<L_SEQ, 256, 0, stream>>>(sp, SPn);

  // Q/K/V = xbf @ Weff + beff  (bf16 out)
  gemm_bf16_kernel<false><<<dim3(DMODEL / 128, L_SEQ / 128), 256, 0, stream>>>(
      xbf, wqe, bqe, Qbf, L_SEQ, DMODEL, DMODEL);
  gemm_bf16_kernel<false><<<dim3(DMODEL / 128, L_SEQ / 128), 256, 0, stream>>>(
      xbf, wke, bke, Kbf, L_SEQ, DMODEL, DMODEL);
  gemm_bf16_kernel<false><<<dim3(DMODEL / 128, L_SEQ / 128), 256, 0, stream>>>(
      xbf, wve, bve, Vbf, L_SEQ, DMODEL, DMODEL);

  // head fastest-varying -> consecutive blocks share SP rows in L2
  flash_attn_kernel<<<dim3(HEADS, L_SEQ / 128), 256, 0, stream>>>(
      Qbf, Kbf, Vbf, SPn, Obf);

  // final projection: f32 output
  gemm_bf16_kernel<true><<<dim3(DMODEL / 128, L_SEQ / 128), 256, 0, stream>>>(
      Obf, owbf, out_b, d_out, L_SEQ, DMODEL, DMODEL);
}
